// HMLSTM_46737834115610
// MI455X (gfx1250) — compile-verified, matrix operand order checked by reading, not run on hardware
//
#include <hip/hip_runtime.h>

// HM-LSTM on gfx1250: bf16 WMMA (v_wmma_f32_16x16x32_bf16) for both the big
// input GEMM and the per-timestep recurrent GEMM; fp32 accumulate & state.
// Step kernel stages h in LDS via async global->LDS copies and computes the
// HM-LSTM boundary gate z with an extra (column-replicated) WMMA tile.
// Recurrent GEMM is spread over 64 blocks (wave = M-tile) for occupancy.

enum { T_ = 512, B_ = 64, I_ = 1024, H_ = 1024, G_ = 4097, GLD_ = 4112 };

typedef __attribute__((ext_vector_type(16))) __bf16 v16bf;
typedef __attribute__((ext_vector_type(8)))  float  v8f;
typedef int vi4 __attribute__((__vector_size__(16)));   // int __vector(4)

#define HAS_ASYNC_LDS (__has_builtin(__builtin_amdgcn_global_load_async_to_lds_b128) && \
                       __has_builtin(__builtin_amdgcn_s_wait_asynccnt))

__device__ __forceinline__ unsigned short f2bf(float f) {
  unsigned int u = __float_as_uint(f);
  u += 0x7FFFu + ((u >> 16) & 1u);          // round-to-nearest-even
  return (unsigned short)(u >> 16);
}
__device__ __forceinline__ float sig_(float x) { return 1.0f / (1.0f + __expf(-x)); }

// 16-bit A-matrix 16x32 lane layout (ISA 7.12.2): lane%16 = row,
// lanes 0-15 hold K={0..7,16..23}, lanes 16-31 hold K={8..15,24..31}.
// Both chunks are contiguous 16B loads. B (K x N) uses the mirrored layout
// with lane%16 = N-column == a row of row-major W, also contiguous in K.
__device__ __forceinline__ v16bf load_frag(const unsigned short* base, int ld,
                                           int row0, int k0) {
  const int lane = threadIdx.x & 31;
  const int r  = row0 + (lane & 15);
  const int kb = (lane & 16) ? 8 : 0;
  const unsigned short* p = base + (size_t)r * ld + (k0 + kb);
  union { v16bf v; uint4 u[2]; } t;
  t.u[0] = *reinterpret_cast<const uint4*>(p);
  t.u[1] = *reinterpret_cast<const uint4*>(p + 16);
  return t.v;
}
__device__ __forceinline__ v16bf load_frag_clamp(const unsigned short* base, int ld,
                                                 int row0, int k0, int rmax) {
  const int lane = threadIdx.x & 31;
  int r = row0 + (lane & 15);
  if (r > rmax) r = rmax;
  const int kb = (lane & 16) ? 8 : 0;
  const unsigned short* p = base + (size_t)r * ld + (k0 + kb);
  union { v16bf v; uint4 u[2]; } t;
  t.u[0] = *reinterpret_cast<const uint4*>(p);
  t.u[1] = *reinterpret_cast<const uint4*>(p + 16);
  return t.v;
}
__device__ __forceinline__ v8f wmma_bf16(v16bf a, v16bf b, v8f c) {
  return __builtin_amdgcn_wmma_f32_16x16x32_bf16(false, a, false, b, (short)0, c,
                                                 false, false);
}

// ---------------------------------------------------------------- init / cvt
__global__ void k_init(unsigned short* __restrict__ h0, float* __restrict__ c) {
  int i = blockIdx.x * blockDim.x + threadIdx.x;
  if (i < B_ * H_) { h0[i] = 0; c[i] = 0.0f; }
}

__global__ void k_cvt_w(const float* __restrict__ wih, const float* __restrict__ whh,
                        const float* __restrict__ bih, const float* __restrict__ bhh,
                        unsigned short* __restrict__ wihb,
                        unsigned short* __restrict__ whhb,
                        float* __restrict__ bias, int n) {
  int i = blockIdx.x * blockDim.x + threadIdx.x;
  if (i < n) { wihb[i] = f2bf(wih[i]); whhb[i] = f2bf(whh[i]); }
  if (i < G_) bias[i] = bih[i] + bhh[i];
}

__global__ void k_cvt_x(const float* __restrict__ x, unsigned short* __restrict__ xb,
                        int n) {
  int i = blockIdx.x * blockDim.x + threadIdx.x;
  if (i < n) xb[i] = f2bf(x[i]);
}

// ------------------------------------------------- phase 1: gx = x @ W_ih^T + b
// grid (M/64, ceil(G/64)), block 128 (4 waves). Wave w -> 64x16 tile.
__global__ __launch_bounds__(128) void k_gemm_gx(
    const unsigned short* __restrict__ xb, const unsigned short* __restrict__ wihb,
    const float* __restrict__ bias, float* __restrict__ gx) {
  const int wave = threadIdx.x >> 5;
  const int lane = threadIdx.x & 31;
  const int m0 = blockIdx.x * 64;
  const int nbase = blockIdx.y * 64 + wave * 16;

  v8f acc[4] = {};
  for (int k0 = 0; k0 < I_; k0 += 32) {
    v16bf b = load_frag_clamp(wihb, I_, nbase, k0, G_ - 1);
#pragma unroll
    for (int mt = 0; mt < 4; ++mt) {
      v16bf a = load_frag(xb, I_, m0 + mt * 16, k0);
      acc[mt] = wmma_bf16(a, b, acc[mt]);
    }
  }

  const int hi8 = (lane & 16) ? 8 : 0;
  const int g = nbase + (lane & 15);
  if (g < G_) {
    const float bv = bias[g];
#pragma unroll
    for (int mt = 0; mt < 4; ++mt)
#pragma unroll
      for (int r = 0; r < 8; ++r) {
        const int m = m0 + mt * 16 + hi8 + r;
        gx[(size_t)m * GLD_ + g] = acc[mt][r] + bv;
      }
  }
}

// ------------------------------------------- phase 2: one recurrent timestep
// grid 64 blocks x 128 threads (4 waves). Block owns 16 h-columns; wave w
// owns M-tile w (16 batch rows) across all 4 gates plus a column-replicated
// z tile -> 5 accumulators per wave, whole gate nonlinearity in-register.
// B-fragments are shared across the block's 4 waves (WGP-cache hits).
__global__ __launch_bounds__(128) void k_step(
    const float* __restrict__ gx_t, const unsigned short* __restrict__ h_in,
    const unsigned short* __restrict__ whhb, float* __restrict__ c,
    unsigned short* __restrict__ h_out, float* __restrict__ out_t,
    float* __restrict__ hT, float* __restrict__ cT, int write_final) {
  extern __shared__ unsigned short lds_h[];   // B_*H_ bf16 = 128 KB (of 320 KB)

  const int tid  = threadIdx.x;
  const int wave = tid >> 5;                  // = M-tile index
  const int lane = tid & 31;
  const int col0 = blockIdx.x * 16;           // 16 h-columns per block
  const int m0   = wave * 16;                 // 16 batch rows per wave

  // ---- stage h (64x1024 bf16) into LDS, async if the toolchain has it
#if HAS_ASYNC_LDS
  {
    unsigned short* hnc = const_cast<unsigned short*>(h_in);
    for (int i = tid; i < (B_ * H_) / 8; i += 128) {
      __builtin_amdgcn_global_load_async_to_lds_b128(
          (__attribute__((address_space(1))) vi4*)(hnc + i * 8),
          (__attribute__((address_space(3))) vi4*)(lds_h + i * 8),
          0, 0);
    }
    __builtin_amdgcn_s_wait_asynccnt(0);
  }
#else
  for (int i = tid; i < (B_ * H_) / 8; i += 128) {
    *reinterpret_cast<uint4*>(lds_h + i * 8) =
        *reinterpret_cast<const uint4*>(h_in + i * 8);
  }
#endif
  __syncthreads();

  // ---- g = h @ W_hh^T : 4 gate tiles + 1 boundary-gate tile (row 4096
  //      clamped across all 16 lanes -> z pre-activation replicated per column)
  v8f acc[4] = {};   // [gate]
  v8f accz  = {};    // z, replicated over columns
  for (int k0 = 0; k0 < H_; k0 += 32) {
    v16bf a = load_frag(lds_h, H_, m0, k0);
    {
      v16bf bz = load_frag_clamp(whhb, H_, G_ - 1, k0, G_ - 1);
      accz = wmma_bf16(a, bz, accz);
    }
#pragma unroll
    for (int gt = 0; gt < 4; ++gt) {
      v16bf b = load_frag(whhb, H_, gt * H_ + col0, k0);
      acc[gt] = wmma_bf16(a, b, acc[gt]);
    }
  }

  // ---- gates + HM-LSTM boundary mixing, all in-register
  const int hi8 = (lane & 16) ? 8 : 0;
  const int col = col0 + (lane & 15);
#pragma unroll
  for (int r = 0; r < 8; ++r) {
    const int m = m0 + hi8 + r;
    const size_t gxrow = (size_t)m * GLD_;
    const float gi = acc[0][r] + gx_t[gxrow + col];
    const float gf = acc[1][r] + gx_t[gxrow + H_ + col];
    const float gc = acc[2][r] + gx_t[gxrow + 2 * H_ + col];
    const float go = acc[3][r] + gx_t[gxrow + 3 * H_ + col];
    const float zv = sig_(accz[r] + gx_t[gxrow + (G_ - 1)]);
    const float iv = sig_(gi), fv = sig_(gf), gv = tanhf(gc), ov = sig_(go);
    const float ig = iv * gv;
    const size_t idx = (size_t)m * H_ + col;
    const float cn = zv * ig + (1.0f - zv) * (fv * c[idx] + ig);
    const float hn = ov * tanhf(cn);
    c[idx] = cn;
    out_t[idx] = hn;
    h_out[idx] = f2bf(hn);
    if (write_final) { hT[idx] = hn; cT[idx] = cn; }
  }
}

// ----------------------------------------------------------------- launcher
extern "C" void kernel_launch(void* const* d_in, const int* in_sizes, int n_in,
                              void* d_out, int out_size, void* d_ws, size_t ws_size,
                              hipStream_t stream) {
  const float* x    = (const float*)d_in[0];
  const float* W_ih = (const float*)d_in[1];
  const float* W_hh = (const float*)d_in[2];
  const float* b_ih = (const float*)d_in[3];
  const float* b_hh = (const float*)d_in[4];

  float* out = (float*)d_out;
  float* hT  = out + (size_t)T_ * B_ * H_;
  float* cT  = hT + (size_t)B_ * H_;

  char* p = (char*)d_ws;
  size_t off = 0;
  auto take = [&](size_t bytes) -> char* {
    char* q = p + off;
    off = (off + bytes + 255) & ~(size_t)255;
    return q;
  };
  unsigned short* xb   = (unsigned short*)take((size_t)T_ * B_ * I_ * 2);
  unsigned short* wihb = (unsigned short*)take((size_t)G_ * I_ * 2);
  unsigned short* whhb = (unsigned short*)take((size_t)G_ * H_ * 2);
  float*          bias = (float*)take((size_t)G_ * 4);
  float*          gx   = (float*)take((size_t)T_ * B_ * GLD_ * 4);
  unsigned short* h0   = (unsigned short*)take((size_t)B_ * H_ * 2);
  unsigned short* h1   = (unsigned short*)take((size_t)B_ * H_ * 2);
  float*          cbuf = (float*)take((size_t)B_ * H_ * 4);

  k_init<<<(B_ * H_ + 255) / 256, 256, 0, stream>>>(h0, cbuf);

  {
    int n = G_ * I_;
    k_cvt_w<<<(n + 255) / 256, 256, 0, stream>>>(W_ih, W_hh, b_ih, b_hh,
                                                 wihb, whhb, bias, n);
  }
  {
    int n = T_ * B_ * I_;
    k_cvt_x<<<(n + 255) / 256, 256, 0, stream>>>(x, xb, n);
  }

  dim3 g1(T_ * B_ / 64, (G_ + 63) / 64);
  k_gemm_gx<<<g1, 128, 0, stream>>>(xb, wihb, bias, gx);

  const size_t step_lds = (size_t)B_ * H_ * sizeof(unsigned short);  // 128 KB
  for (int t = 0; t < T_; ++t) {
    const unsigned short* hin = (t & 1) ? h1 : h0;
    unsigned short*      hout = (t & 1) ? h0 : h1;
    k_step<<<H_ / 16, 128, step_lds, stream>>>(
        gx + (size_t)t * B_ * GLD_, hin, whhb, cbuf,
        hout, out + (size_t)t * B_ * H_,
        hT, cT, (t == T_ - 1) ? 1 : 0);
  }
}
